// CfCBlock_74947179315325
// MI455X (gfx1250) — matrix operand order repaired and checked by reading, not compiled, not run on hardware
//
#include <hip/hip_runtime.h>
#include <hip/hip_bf16.h>
#include <cstdint>
#include <cstddef>

typedef __bf16 bf16;
typedef __bf16 v8bf  __attribute__((ext_vector_type(8)));
typedef __bf16 v16bf __attribute__((ext_vector_type(16)));
typedef float  v8f   __attribute__((ext_vector_type(8)));

#define BB     32
#define LSEQ   1024
#define DMODEL 512

// Per-NCP-layer geometry (inter, command, motor). K padded to 32, N padded to 16.
constexpr int IN_[3]    = {512, 308, 204};   // feed-forward input width
constexpr int H_[3]     = {308, 204, 512};   // hidden width
constexpr int INPAD_[3] = {512, 320, 224};   // padK(IN)
constexpr int HK_[3]    = {320, 224, 512};   // padK(H)  (row stride of h buffers)
constexpr int CATP_[3]  = {832, 544, 736};   // INPAD + HK
constexpr int KT_[3]    = {26, 17, 23};      // CATP / 32
constexpr int NT_[3]    = {20, 13, 32};      // ceil(H/16)
constexpr int PADN_[3]  = {320, 208, 512};   // NT*16
constexpr int NWG_TOT   = 2*20 + 2*13 + 2*32;      // 130 workgroups (mtiles x ntiles)
constexpr int SMEM_BYTES = 4 * 16 * 832 * 2;       // 106496 B: worst-case weight slice

// ---------------------------------------------------------------- utility kernels

__global__ void zero_kernel(unsigned* __restrict__ p, size_t n) {
  size_t i = (size_t)blockIdx.x * 256 + threadIdx.x;
  if (i < n) p[i] = 0u;
}

__global__ void init_h_kernel(const float* __restrict__ hx,
                              float* __restrict__ h0f, float* __restrict__ h1f, float* __restrict__ h2f,
                              bf16* __restrict__ h0b, bf16* __restrict__ h1b, bf16* __restrict__ h2b) {
  int idx = blockIdx.x * 256 + threadIdx.x;
  if (idx >= BB * 1024) return;
  int b = idx >> 10, j = idx & 1023;
  float v = hx[idx];
  if (j < 308)      { h0f[b*320 + j]       = v; h0b[b*320 + j]       = (bf16)v; }
  else if (j < 512) { h1f[b*224 + (j-308)] = v; h1b[b*224 + (j-308)] = (bf16)v; }
  else              { h2f[b*512 + (j-512)] = v; h2b[b*512 + (j-512)] = (bf16)v; }
}

__global__ void pack_hn_kernel(const float* __restrict__ h0, const float* __restrict__ h1,
                               const float* __restrict__ h2, float* __restrict__ dst) {
  int idx = blockIdx.x * 256 + threadIdx.x;
  if (idx >= BB * 1024) return;
  int b = idx >> 10, j = idx & 1023;
  float v;
  if (j < 308)      v = h0[b*320 + j];
  else if (j < 512) v = h1[b*224 + (j-308)];
  else              v = h2[b*512 + (j-512)];
  dst[idx] = v;
}

// LayerNorm over D=512, emit bf16 activations for WMMA A-operands.
__global__ void ln_kernel(const float* __restrict__ x, const float* __restrict__ g,
                          const float* __restrict__ b, bf16* __restrict__ xn) {
  __shared__ float ssum[256];
  __shared__ float ssq[256];
  const int row = blockIdx.x;          // b*L + t
  const int tid = threadIdx.x;         // 256 threads, 2 elems each
  const float* xr = x + (size_t)row * DMODEL;
  float v0 = xr[tid], v1 = xr[tid + 256];
  ssum[tid] = v0 + v1;
  ssq[tid]  = v0 * v0 + v1 * v1;
  __syncthreads();
  for (int s = 128; s > 0; s >>= 1) {
    if (tid < s) { ssum[tid] += ssum[tid + s]; ssq[tid] += ssq[tid + s]; }
    __syncthreads();
  }
  const float mu  = ssum[0] * (1.f / DMODEL);
  const float var = ssq[0] * (1.f / DMODEL) - mu * mu;
  const float rs  = rsqrtf(var + 1e-5f);
  bf16* xo = xn + (size_t)row * DMODEL;
  xo[tid]       = (bf16)((v0 - mu) * rs * g[tid]       + b[tid]);
  xo[tid + 256] = (bf16)((v1 - mu) * rs * g[tid + 256] + b[tid + 256]);
}

// Pack one weight matrix (h x (in+h), row-major) into padded bf16 (padN x catp),
// applying the NCP wiring mask (ff1/ff2 only) and zero-filling K/N padding.
__global__ void pack_w_kernel(const float* __restrict__ w, const float* __restrict__ mask,
                              bf16* __restrict__ dst,
                              int h, int in_l, int inpad, int padN, int catp) {
  int idx = blockIdx.x * 256 + threadIdx.x;
  int total = padN * catp;
  if (idx >= total) return;
  int n = idx / catp, k = idx % catp;
  float v = 0.f;
  if (n < h) {
    int cat = in_l + h;
    int sk = -1;
    if (k < in_l)                          sk = k;
    else if (k >= inpad && k < inpad + h)  sk = in_l + (k - inpad);
    if (sk >= 0) {
      v = w[(size_t)n * cat + sk];
      if (mask) v *= mask[(size_t)n * cat + sk];
    }
  }
  dst[idx] = (bf16)v;
}

// ---------------------------------------------------------------- persistent scan

struct ScanArgs {
  const bf16*  xn;         // (B, L, 512) bf16
  const bf16*  w[12];      // [layer*4 + q], q: ff1,ff2,ta,tb ; padN x catp bf16
  const float* bias[12];
  bf16*        hb[6];      // [layer*2 + parity], B x HK bf16 (double buffered)
  float*       hf[3];      // B x HK f32 (latest state, for hn)
  const float* x;          // original input for residual
  float*       out;        // (B, L, 512) f32
  unsigned*    bar_cnt;
  unsigned*    bar_gen;
};

__launch_bounds__(32, 1)
__global__ void cfc_scan_kernel(ScanArgs a) {
  extern __shared__ __align__(64) unsigned char smem_raw[];
  bf16* wl = (bf16*)smem_raw;

  const int lane = threadIdx.x;
  const int bid  = blockIdx.x;
  int layer, mt, nt;
  if (bid < 40)      { layer = 0; mt = bid / 20;            nt = bid % 20; }
  else if (bid < 66) { layer = 1; int i = bid - 40; mt = i / 13; nt = i % 13; }
  else               { layer = 2; int i = bid - 66; mt = i / 32; nt = i % 32; }

  const int h      = H_[layer];
  const int inpad  = INPAD_[layer];
  const int hk     = HK_[layer];
  const int catp   = CATP_[layer];
  const int ktiles = KT_[layer];

  // Stage this tile's weight slice into LDS once: 4 matrices x 16 rows x catp bf16.
  for (int q = 0; q < 4; ++q) {
    const uint4* src = (const uint4*)(a.w[layer*4 + q] + (size_t)(nt * 16) * catp);
    uint4* dst = (uint4*)(wl + (size_t)q * 16 * catp);
    const int nchunks = (16 * catp) / 8;       // 8 bf16 per 16B
    for (int i = lane; i < nchunks; i += 32) dst[i] = src[i];
  }
  __syncthreads();

  const int half = lane >> 4;    // ISA A/B fragment half-wave selector
  const int l15  = lane & 15;
  const int arow = mt * 16 + l15;  // batch row (A rows: lanes 0-15 and 16-31 both M=0..15)
  const int ncol = nt * 16 + l15;  // output column

  float b1 = 0.f, b2 = 0.f, b3 = 0.f, b4 = 0.f;
  if (ncol < h) {
    b1 = a.bias[layer*4 + 0][ncol];
    b2 = a.bias[layer*4 + 1][ncol];
    b3 = a.bias[layer*4 + 2][ncol];
    b4 = a.bias[layer*4 + 3][ncol];
  }

  unsigned myGen = 0;

  for (int t = 0; t < LSEQ; ++t) {
    const int rp = t & 1, wp = rp ^ 1;   // read h(t-1) from rp, write h(t) to wp
    for (int stage = 0; stage < 3; ++stage) {
      if (layer == stage) {
        const bf16* arow_in;
        if (layer == 0) arow_in = a.xn + ((size_t)arow * LSEQ + t) * DMODEL;
        else            arow_in = a.hb[(layer - 1) * 2 + wp] + (size_t)arow * HK_[layer - 1];
        const bf16* arow_h = a.hb[layer * 2 + rp] + (size_t)arow * hk;

        v8f acc0 = {0.f,0.f,0.f,0.f,0.f,0.f,0.f,0.f};
        v8f acc1 = acc0, acc2 = acc0, acc3 = acc0;

        for (int kt = 0; kt < ktiles; ++kt) {
          const int k0 = kt * 32;
          // A fragment: 16x32 bf16; lane holds row l15, K = half*16..+15 (contiguous)
          const bf16* ap = (k0 < inpad) ? (arow_in + k0 + half * 16)
                                        : (arow_h + (k0 - inpad) + half * 16);
          v8bf lo = *(const v8bf*)(ap);
          v8bf hi = *(const v8bf*)(ap + 8);
          v16bf av = __builtin_shufflevector(lo, hi,
              0,1,2,3,4,5,6,7,8,9,10,11,12,13,14,15);
          // B fragment: 32x16 (= W^T); lane holds weight row n=l15, K = half*16..+15 (contiguous)
          const bf16* lb = wl + (size_t)l15 * catp + k0 + half * 16;
          v16bf bv0 = *(const v16bf*)(lb + 0 * 16 * catp);
          v16bf bv1 = *(const v16bf*)(lb + 1 * 16 * catp);
          v16bf bv2 = *(const v16bf*)(lb + 2 * 16 * catp);
          v16bf bv3 = *(const v16bf*)(lb + 3 * 16 * catp);
          acc0 = __builtin_amdgcn_wmma_f32_16x16x32_bf16(false, av, false, bv0, (short)0, acc0, false, false);
          acc1 = __builtin_amdgcn_wmma_f32_16x16x32_bf16(false, av, false, bv1, (short)0, acc1, false, false);
          acc2 = __builtin_amdgcn_wmma_f32_16x16x32_bf16(false, av, false, bv2, (short)0, acc2, false, false);
          acc3 = __builtin_amdgcn_wmma_f32_16x16x32_bf16(false, av, false, bv3, (short)0, acc3, false, false);
        }

        if (ncol < h) {
          float* hfw = a.hf[layer];
          bf16*  hbw = a.hb[layer * 2 + wp];
          #pragma unroll
          for (int r = 0; r < 8; ++r) {
            const int row = mt * 16 + half * 8 + r;     // C/D: lanes>=16 carry M=8..15
            float f1 = tanhf(acc0[r] + b1);
            float f2 = tanhf(acc1[r] + b2);
            float z  = (acc2[r] + b3) + (acc3[r] + b4);
            float ti = 1.f / (1.f + __expf(-z));
            float hv = f1 * (1.f - ti) + ti * f2;
            hfw[(size_t)row * hk + ncol] = hv;
            hbw[(size_t)row * hk + ncol] = (bf16)hv;
            if (layer == 2) {   // motor layer: residual output
              size_t xi = ((size_t)row * LSEQ + t) * DMODEL + ncol;
              a.out[xi] = a.x[xi] + hv;
            }
          }
        }
      }
      // ---- grid-wide barrier between stages (sense via monotone generation) ----
      ++myGen;
      __threadfence();
      if (lane == 0) {
        unsigned old = __hip_atomic_fetch_add(a.bar_cnt, 1u, __ATOMIC_ACQ_REL, __HIP_MEMORY_SCOPE_AGENT);
        if (old == (unsigned)(NWG_TOT - 1)) {
          __hip_atomic_store(a.bar_cnt, 0u, __ATOMIC_RELAXED, __HIP_MEMORY_SCOPE_AGENT);
          __hip_atomic_fetch_add(a.bar_gen, 1u, __ATOMIC_RELEASE, __HIP_MEMORY_SCOPE_AGENT);
        }
      }
      while (__hip_atomic_load(a.bar_gen, __ATOMIC_ACQUIRE, __HIP_MEMORY_SCOPE_AGENT) < myGen)
        __builtin_amdgcn_s_sleep(1);
      __threadfence();
    }
  }
}

// ---------------------------------------------------------------- host launcher

extern "C" void kernel_launch(void* const* d_in, const int* in_sizes, int n_in,
                              void* d_out, int out_size, void* d_ws, size_t ws_size,
                              hipStream_t stream) {
  (void)in_sizes; (void)n_in; (void)out_size; (void)ws_size;
  const float* x    = (const float*)d_in[0];
  const float* hx   = (const float*)d_in[1];
  const float* ln_g = (const float*)d_in[2];
  const float* ln_b = (const float*)d_in[3];
  const float *W[12], *Bv[12], *M3[3];
  for (int l = 0; l < 3; ++l) {
    W[l*4+0]  = (const float*)d_in[4  + l];   // ff1_w
    W[l*4+1]  = (const float*)d_in[10 + l];   // ff2_w
    W[l*4+2]  = (const float*)d_in[16 + l];   // ta_w
    W[l*4+3]  = (const float*)d_in[22 + l];   // tb_w
    Bv[l*4+0] = (const float*)d_in[7  + l];   // ff1_b
    Bv[l*4+1] = (const float*)d_in[13 + l];   // ff2_b
    Bv[l*4+2] = (const float*)d_in[19 + l];   // ta_b
    Bv[l*4+3] = (const float*)d_in[25 + l];   // tb_b
    M3[l]     = (const float*)d_in[28 + l];   // masks
  }

  char* ws = (char*)d_ws;
  size_t cur = 0;
  auto carve = [&](size_t bytes) -> char* {
    char* p = ws + cur;
    cur += (bytes + 255) & ~size_t(255);
    return p;
  };
  bf16* xn = (bf16*)carve((size_t)BB * LSEQ * DMODEL * 2);
  bf16* wbuf[12];
  for (int l = 0; l < 3; ++l)
    for (int q = 0; q < 4; ++q)
      wbuf[l*4+q] = (bf16*)carve((size_t)PADN_[l] * CATP_[l] * 2);
  size_t zstart = cur;                  // everything below gets zeroed each call
  bf16* hb[6];
  for (int l = 0; l < 3; ++l)
    for (int p = 0; p < 2; ++p)
      hb[l*2+p] = (bf16*)carve((size_t)BB * HK_[l] * 2);
  float* hf[3];
  for (int l = 0; l < 3; ++l) hf[l] = (float*)carve((size_t)BB * HK_[l] * 4);
  unsigned* bar = (unsigned*)carve(256);
  size_t zbytes = cur - zstart;

  // 1) zero hidden-state buffers (incl. padding) + barrier counters
  {
    size_t nwords = zbytes / 4;
    zero_kernel<<<(int)((nwords + 255) / 256), 256, 0, stream>>>((unsigned*)(ws + zstart), nwords);
  }
  // 2) scatter hx into parity-0 state buffers
  init_h_kernel<<<(BB * 1024 + 255) / 256, 256, 0, stream>>>(hx, hf[0], hf[1], hf[2],
                                                             hb[0], hb[2], hb[4]);
  // 3) LayerNorm -> bf16 activations
  ln_kernel<<<BB * LSEQ, 256, 0, stream>>>(x, ln_g, ln_b, xn);
  // 4) pack + mask + pad weights to bf16 (done once per call; weights ~5.8 MB bf16)
  for (int l = 0; l < 3; ++l) {
    int total  = PADN_[l] * CATP_[l];
    int blocks = (total + 255) / 256;
    for (int q = 0; q < 4; ++q) {
      const float* mk = (q < 2) ? M3[l] : nullptr;   // mask only ff1/ff2
      pack_w_kernel<<<blocks, 256, 0, stream>>>(W[l*4+q], mk, wbuf[l*4+q],
                                                H_[l], IN_[l], INPAD_[l], PADN_[l], CATP_[l]);
    }
  }
  // 5) persistent recurrent scan: weights live in LDS, WMMA bf16, grid barriers
  ScanArgs sa{};
  sa.xn = xn;
  for (int i = 0; i < 12; ++i) { sa.w[i] = wbuf[i]; sa.bias[i] = Bv[i]; }
  for (int i = 0; i < 6;  ++i) sa.hb[i] = hb[i];
  for (int i = 0; i < 3;  ++i) sa.hf[i] = hf[i];
  sa.x = x; sa.out = (float*)d_out;
  sa.bar_cnt = bar; sa.bar_gen = bar + 1;
  (void)hipFuncSetAttribute((const void*)cfc_scan_kernel,
                            hipFuncAttributeMaxDynamicSharedMemorySize, SMEM_BYTES);
  cfc_scan_kernel<<<NWG_TOT, 32, SMEM_BYTES, stream>>>(sa);
  // 6) final hidden state -> tail of d_out
  pack_hn_kernel<<<(BB * 1024 + 255) / 256, 256, 0, stream>>>(
      hf[0], hf[1], hf[2], (float*)d_out + (size_t)BB * LSEQ * DMODEL);
}
